// AttDecoder_66202625900750
// MI455X (gfx1250) — matrix operand order repaired and checked by reading, not compiled, not run on hardware
//
#include <hip/hip_runtime.h>
#include <hip/hip_bf16.h>

// ---------------- problem constants (from reference) ----------------
#define BB   128
#define TT   20
#define LL   196
#define EE   512
#define HH   512
#define AA   512
#define VV   10000
#define FEA  2048

typedef __attribute__((ext_vector_type(16))) __bf16          v16bf;
typedef __attribute__((ext_vector_type(8)))  __bf16          v8bf;
typedef __attribute__((ext_vector_type(8)))  float           v8f;

__device__ __forceinline__ unsigned short f2bf(float f) {
    unsigned int u = __float_as_uint(f);
    unsigned int r = u + 0x7FFFu + ((u >> 16) & 1u);   // round-to-nearest-even
    return (unsigned short)(r >> 16);
}

__device__ __forceinline__ float sigmoidf_(float x) {
    return 1.0f / (1.0f + __expf(-x));
}

// ------------------------------------------------------------------
// f32 -> bf16 conversion (weights: once per launch; activations: per step)
// ------------------------------------------------------------------
__global__ __launch_bounds__(256)
void f32_to_bf16_kernel(const float* __restrict__ src,
                        unsigned short* __restrict__ dst, int n)
{
    int i = blockIdx.x * 256 + threadIdx.x;
    if (i < n) dst[i] = f2bf(src[i]);
}

// ------------------------------------------------------------------
// Fragment bundle for one 32-deep K chunk of a 32M x 32N wave tile.
// ------------------------------------------------------------------
struct Frags {
    v16bf a0, a1;   // A rows  [m, m+16]    (16-bit A 16x32 ISA layout)
    v16bf b0, b1;   // B tiles [n0, n0+16]  (16-bit B 32x16 ISA layout)
};

__device__ __forceinline__ v16bf load_a_frag(const __bf16* Arow, int kA)
{
    v8bf lo = *(const v8bf*)(Arow + kA);
    v8bf hi = *(const v8bf*)(Arow + kA + 16);
    return __builtin_shufflevector(lo, hi, 0,1,2,3,4,5,6,7,8,9,10,11,12,13,14,15);
}

__device__ __forceinline__ v16bf load_b_frag(const __bf16* Wrow, int kB)
{
    v8bf lo = *(const v8bf*)(Wrow + kB);
    v8bf hi = *(const v8bf*)(Wrow + kB + 16);
    return __builtin_shufflevector(lo, hi, 0,1,2,3,4,5,6,7,8,9,10,11,12,13,14,15);
}

__device__ __forceinline__ Frags load_frags(const __bf16* Arow0, const __bf16* Arow1,
                                            const __bf16* Wrow0, const __bf16* Wrow1,
                                            int k0, int sel)
{
    Frags f;
    const int kA = k0 + sel * 8;    // lanes 0-15: K k0..+7 / k0+16..+23; 16-31: +8
    const int kB = k0 + sel * 8;    // same K order as A: k0..+7 / k0+16..+23; 16-31: +8
    f.a0 = load_a_frag(Arow0, kA);
    f.a1 = load_a_frag(Arow1, kA);
    f.b0 = load_b_frag(Wrow0, kB);
    f.b1 = load_b_frag(Wrow1, kB);
    return f;
}

// ------------------------------------------------------------------
// GEMM:  C[M,N] = A[M,K](bf16) * W[N,K](bf16)^T (+ bias) (+= existing C)
// WMMA bf16 16x16x32, f32 accumulate, software-pipelined K loop.
// Block = 128 threads = 4 waves. Each wave: 32(M) x 32(N) register tile
// (2x2 accumulators). Block tile = 32M x 128N.
// Requires M % 32 == 0, K % 32 == 0; N % 16 == 0 (wave-uniform N tail).
// ------------------------------------------------------------------
__global__ __launch_bounds__(128)
void gemm_wmma_bf16(const __bf16* __restrict__ A, int lda,
                    const __bf16* __restrict__ W, int ldw,
                    const float* __restrict__ bias,
                    float* __restrict__ C, long long ldc,
                    int N, int K, int accumulate)
{
    const int wave   = threadIdx.x >> 5;
    const int lane   = threadIdx.x & 31;
    const int mTile  = blockIdx.y * 32;
    const int nTile0 = blockIdx.x * 128 + wave * 32;
    if (nTile0 >= N) return;                 // wave-uniform: EXEC stays all-1s
    const int  nTile1 = nTile0 + 16;
    const bool hasN1  = (nTile1 < N);        // wave-uniform

    const int m   = lane & 15;               // row within 16-tile / col index n
    const int sel = lane >> 4;               // K-half selector per ISA layout
    const int n   = m;

    const __bf16* __restrict__ Arow0 = A + (size_t)(mTile + m) * (size_t)lda;
    const __bf16* __restrict__ Arow1 = Arow0 + (size_t)16 * (size_t)lda;
    const __bf16* __restrict__ Wrow0 = W + (size_t)(nTile0 + n) * (size_t)ldw;
    const __bf16* __restrict__ Wrow1 = hasN1 ? (W + (size_t)(nTile1 + n) * (size_t)ldw)
                                             : Wrow0;   // safe dummy, in-bounds

    v8f acc00 = {}, acc10 = {}, acc01 = {}, acc11 = {};
    const int mBase = sel * 8;
    if (accumulate) {
        #pragma unroll
        for (int r = 0; r < 8; ++r) {
            acc00[r] = C[(size_t)(mTile + mBase + r)      * (size_t)ldc + (nTile0 + n)];
            acc10[r] = C[(size_t)(mTile + 16 + mBase + r) * (size_t)ldc + (nTile0 + n)];
        }
        if (hasN1) {
            #pragma unroll
            for (int r = 0; r < 8; ++r) {
                acc01[r] = C[(size_t)(mTile + mBase + r)      * (size_t)ldc + (nTile1 + n)];
                acc11[r] = C[(size_t)(mTile + 16 + mBase + r) * (size_t)ldc + (nTile1 + n)];
            }
        }
    }

    // ---- software-pipelined K loop: load chunk k+32 under WMMAs on chunk k ----
    Frags cur = load_frags(Arow0, Arow1, Wrow0, Wrow1, 0, sel);
    int k0 = 0;
    for (; k0 + 32 < K; k0 += 32) {
        __builtin_prefetch(Wrow0 + k0 + 64, 0, 1);      // global_prefetch_b8
        Frags nxt = load_frags(Arow0, Arow1, Wrow0, Wrow1, k0 + 32, sel);
        acc00 = __builtin_amdgcn_wmma_f32_16x16x32_bf16(
                    false, cur.a0, false, cur.b0, (short)0, acc00, false, false);
        acc10 = __builtin_amdgcn_wmma_f32_16x16x32_bf16(
                    false, cur.a1, false, cur.b0, (short)0, acc10, false, false);
        if (hasN1) {
            acc01 = __builtin_amdgcn_wmma_f32_16x16x32_bf16(
                        false, cur.a0, false, cur.b1, (short)0, acc01, false, false);
            acc11 = __builtin_amdgcn_wmma_f32_16x16x32_bf16(
                        false, cur.a1, false, cur.b1, (short)0, acc11, false, false);
        }
        cur = nxt;
    }
    acc00 = __builtin_amdgcn_wmma_f32_16x16x32_bf16(
                false, cur.a0, false, cur.b0, (short)0, acc00, false, false);
    acc10 = __builtin_amdgcn_wmma_f32_16x16x32_bf16(
                false, cur.a1, false, cur.b0, (short)0, acc10, false, false);
    if (hasN1) {
        acc01 = __builtin_amdgcn_wmma_f32_16x16x32_bf16(
                    false, cur.a0, false, cur.b1, (short)0, acc01, false, false);
        acc11 = __builtin_amdgcn_wmma_f32_16x16x32_bf16(
                    false, cur.a1, false, cur.b1, (short)0, acc11, false, false);
    }

    // C/D 16x16 f32 layout: lanes 0-15 -> M=r, lanes 16-31 -> M=r+8, N=n
    const float bv0 = bias ? bias[nTile0 + n] : 0.0f;
    #pragma unroll
    for (int r = 0; r < 8; ++r) {
        C[(size_t)(mTile + mBase + r)      * (size_t)ldc + (nTile0 + n)] = acc00[r] + bv0;
        C[(size_t)(mTile + 16 + mBase + r) * (size_t)ldc + (nTile0 + n)] = acc10[r] + bv0;
    }
    if (hasN1) {
        const float bv1 = bias ? bias[nTile1 + n] : 0.0f;
        #pragma unroll
        for (int r = 0; r < 8; ++r) {
            C[(size_t)(mTile + mBase + r)      * (size_t)ldc + (nTile1 + n)] = acc01[r] + bv1;
            C[(size_t)(mTile + 16 + mBase + r) * (size_t)ldc + (nTile1 + n)] = acc11[r] + bv1;
        }
    }
}

// ------------------------------------------------------------------
// Fused attention step: att scores + softmax + context z + beta gate.
// One block (256 threads = 8 waves) per batch element.
// ------------------------------------------------------------------
__global__ __launch_bounds__(256)
void att_step_kernel(const float* __restrict__ fea_att,   // [B,L,A]
                     const float* __restrict__ fea_maps,  // [B,L,E]
                     const float* __restrict__ h,         // [B,H] (f32)
                     const float* __restrict__ hid_att,   // [B,A] (precomputed)
                     const float* __restrict__ att_w,     // [A]
                     const float* __restrict__ att_b,     // [1]
                     const float* __restrict__ fb_w,      // [H]
                     const float* __restrict__ fb_b,      // [1]
                     float* __restrict__ alphas_out,      // [B,T,L]
                     float* __restrict__ betas_out,       // [B,T]
                     float* __restrict__ z,               // [B,E]
                     int t)
{
    __shared__ float sHid[AA];
    __shared__ float sAw[AA];
    __shared__ float sAtt[224];
    __shared__ float sRed[32];

    const int tid  = threadIdx.x;
    const int b    = blockIdx.x;
    const int wave = tid >> 5;
    const int lane = tid & 31;

    for (int i = tid; i < AA; i += 256) {
        sHid[i] = hid_att[b * AA + i];
        sAw[i]  = att_w[i];
    }
    __syncthreads();

    const float* fa = fea_att + (size_t)b * LL * AA;
    const float attb = att_b[0];
    for (int l = wave; l < LL; l += 8) {
        const float* row = fa + (size_t)l * AA;
        float acc = 0.0f;
        for (int a = lane; a < AA; a += 32) {
            float v = row[a] + sHid[a];
            v = v > 0.0f ? v : 0.0f;
            acc += v * sAw[a];
        }
        #pragma unroll
        for (int off = 16; off > 0; off >>= 1) acc += __shfl_down(acc, off, 32);
        if (lane == 0) sAtt[l] = acc + attb;
    }
    __syncthreads();

    if (tid == 0) {
        float mx = -3.4e38f;
        for (int l = 0; l < LL; ++l) mx = fmaxf(mx, sAtt[l]);
        float s = 0.0f;
        for (int l = 0; l < LL; ++l) { float e = __expf(sAtt[l] - mx); sAtt[l] = e; s += e; }
        sRed[0] = 1.0f / s;
    }
    __syncthreads();
    const float inv = sRed[0];
    for (int l = tid; l < LL; l += 256) {
        float al = sAtt[l] * inv;
        sAtt[l] = al;
        alphas_out[((size_t)b * TT + t) * LL + l] = al;
    }

    float p = 0.0f;
    for (int i = tid; i < HH; i += 256) p += h[b * HH + i] * fb_w[i];
    #pragma unroll
    for (int off = 16; off > 0; off >>= 1) p += __shfl_down(p, off, 32);
    if (lane == 0) sRed[8 + wave] = p;
    __syncthreads();
    if (tid == 0) {
        float s = 0.0f;
        for (int w = 0; w < 8; ++w) s += sRed[8 + w];
        float beta = sigmoidf_(s + fb_b[0]);
        sRed[1] = beta;
        betas_out[(size_t)b * TT + t] = beta;
    }
    __syncthreads();
    const float beta = sRed[1];

    const float* fm = fea_maps + (size_t)b * LL * EE;
    for (int e = tid; e < EE; e += 256) {
        float acc = 0.0f;
        for (int l = 0; l < LL; ++l) acc += sAtt[l] * fm[(size_t)l * EE + e];
        z[(size_t)b * EE + e] = beta * acc;
    }
}

// x_bf[b, 0:E] = bf16(embed_w[cap[b,t]]);  x_bf[b, E:2E] = bf16(z[b])
__global__ __launch_bounds__(256)
void build_x_kernel(const float* __restrict__ embed_w,
                    const int* __restrict__ cap,
                    const float* __restrict__ z,
                    unsigned short* __restrict__ x_bf, int t)
{
    int idx = blockIdx.x * 256 + threadIdx.x;      // B * 2E
    if (idx >= BB * 2 * EE) return;
    int b = idx >> 10, col = idx & 1023;
    float v;
    if (col < EE) {
        int tok = cap[b * TT + t];
        v = embed_w[(size_t)tok * EE + col];
    } else {
        v = z[(size_t)b * EE + (col - EE)];
    }
    x_bf[idx] = f2bf(v);
}

// LSTM pointwise update (gate order i,f,g,o); writes h in f32 + bf16
__global__ __launch_bounds__(256)
void lstm_cell_kernel(const float* __restrict__ gates,   // [B,4H]
                      float* __restrict__ h, float* __restrict__ c,
                      unsigned short* __restrict__ h_bf)
{
    int idx = blockIdx.x * 256 + threadIdx.x;      // B*H
    if (idx >= BB * HH) return;
    int b = idx >> 9, j = idx & 511;
    const float* g = gates + (size_t)b * 4 * HH;
    float ig = sigmoidf_(g[j]);
    float fg = sigmoidf_(g[HH + j]);
    float gg = tanhf(g[2 * HH + j]);
    float og = sigmoidf_(g[3 * HH + j]);
    float cn = fg * c[idx] + ig * gg;
    float hn = og * tanhf(cn);
    c[idx] = cn;
    h[idx] = hn;
    h_bf[idx] = f2bf(hn);
}

// ------------------------------------------------------------------
static inline void gemm(const unsigned short* A, int lda,
                        const unsigned short* W, int ldw,
                        const float* bias, float* C, long long ldc,
                        int M, int N, int K, int accumulate, hipStream_t s)
{
    dim3 grid((N + 127) / 128, M / 32);
    gemm_wmma_bf16<<<grid, 128, 0, s>>>((const __bf16*)A, lda, (const __bf16*)W,
                                        ldw, bias, C, ldc, N, K, accumulate);
}

static inline void convert(const float* src, unsigned short* dst, int n, hipStream_t s)
{
    f32_to_bf16_kernel<<<(n + 255) / 256, 256, 0, s>>>(src, dst, n);
}

extern "C" void kernel_launch(void* const* d_in, const int* in_sizes, int n_in,
                              void* d_out, int out_size, void* d_ws, size_t ws_size,
                              hipStream_t stream)
{
    const float* fea_vec   = (const float*)d_in[0];
    const float* fea_maps  = (const float*)d_in[1];
    const int*   cap       = (const int*)  d_in[2];
    // d_in[3] = cap_len (== TT, compile-time constant here)
    const float* embed_w   = (const float*)d_in[4];
    const float* W_ih      = (const float*)d_in[5];
    const float* b_ih      = (const float*)d_in[6];
    const float* W_hh      = (const float*)d_in[7];
    const float* b_hh      = (const float*)d_in[8];
    const float* fc_w      = (const float*)d_in[9];
    const float* fc_b      = (const float*)d_in[10];
    const float* fea_att_w = (const float*)d_in[11];
    const float* fea_att_b = (const float*)d_in[12];
    const float* hid_att_w = (const float*)d_in[13];
    const float* hid_att_b = (const float*)d_in[14];
    const float* att_w     = (const float*)d_in[15];
    const float* att_b     = (const float*)d_in[16];
    const float* init_h_w  = (const float*)d_in[17];
    const float* init_h_b  = (const float*)d_in[18];
    const float* init_c_w  = (const float*)d_in[19];
    const float* init_c_b  = (const float*)d_in[20];
    const float* fb_w      = (const float*)d_in[21];
    const float* fb_b      = (const float*)d_in[22];

    float* outw = (float*)d_out;                               // [B,T,V]
    float* outa = outw + (size_t)BB * TT * VV;                 // [B,T,L]
    float* outb = outa + (size_t)BB * TT * LL;                 // [B,T,1]

    // ---- workspace layout ----
    float* wsf     = (float*)d_ws;
    float* h       = wsf;  wsf += (size_t)BB * HH;
    float* c       = wsf;  wsf += (size_t)BB * HH;
    float* hid_att = wsf;  wsf += (size_t)BB * AA;
    float* z       = wsf;  wsf += (size_t)BB * EE;
    float* gates   = wsf;  wsf += (size_t)BB * 4 * HH;
    float* fea_att = wsf;  wsf += (size_t)BB * LL * AA;        // [B*L, A]

    unsigned short* bfp        = (unsigned short*)wsf;
    unsigned short* h_bf       = bfp;  bfp += (size_t)BB * HH;
    unsigned short* x_bf       = bfp;  bfp += (size_t)BB * 2 * EE;
    unsigned short* fea_vec_bf = bfp;  bfp += (size_t)BB * FEA;
    unsigned short* fmaps_bf   = bfp;  bfp += (size_t)BB * LL * EE;
    unsigned short* ihw_bf     = bfp;  bfp += (size_t)HH * FEA;
    unsigned short* icw_bf     = bfp;  bfp += (size_t)HH * FEA;
    unsigned short* faw_bf     = bfp;  bfp += (size_t)AA * EE;
    unsigned short* haw_bf     = bfp;  bfp += (size_t)AA * HH;
    unsigned short* wih_bf     = bfp;  bfp += (size_t)4 * HH * 2 * EE;
    unsigned short* whh_bf     = bfp;  bfp += (size_t)4 * HH * HH;
    unsigned short* fcw_bf     = bfp;  bfp += (size_t)VV * HH;

    // ---- one-time bf16 conversion of GEMM operands ----
    convert(fea_vec,   fea_vec_bf, BB * FEA, stream);
    convert(fea_maps,  fmaps_bf,   BB * LL * EE, stream);
    convert(init_h_w,  ihw_bf,     HH * FEA, stream);
    convert(init_c_w,  icw_bf,     HH * FEA, stream);
    convert(fea_att_w, faw_bf,     AA * EE, stream);
    convert(hid_att_w, haw_bf,     AA * HH, stream);
    convert(W_ih,      wih_bf,     4 * HH * 2 * EE, stream);
    convert(W_hh,      whh_bf,     4 * HH * HH, stream);
    convert(fc_w,      fcw_bf,     VV * HH, stream);

    // ---- one-time precompute ----
    gemm(fea_vec_bf, FEA, ihw_bf, FEA, init_h_b, h, HH, BB, HH, FEA, 0, stream);
    gemm(fea_vec_bf, FEA, icw_bf, FEA, init_c_b, c, HH, BB, HH, FEA, 0, stream);
    convert(h, h_bf, BB * HH, stream);
    gemm(fmaps_bf, EE, faw_bf, EE, fea_att_b, fea_att, AA, BB * LL, AA, EE, 0, stream);

    // ---- recurrent decode loop ----
    for (int t = 0; t < TT; ++t) {
        gemm(h_bf, HH, haw_bf, HH, hid_att_b, hid_att, AA, BB, AA, HH, 0, stream);

        att_step_kernel<<<BB, 256, 0, stream>>>(fea_att, fea_maps, h, hid_att,
                                                att_w, att_b, fb_w, fb_b,
                                                outa, outb, z, t);

        build_x_kernel<<<(BB * 2 * EE) / 256, 256, 0, stream>>>(embed_w, cap, z, x_bf, t);

        // gates = x @ W_ih^T + b_ih  (K = 2E = 1024)
        gemm(x_bf, 2 * EE, wih_bf, 2 * EE, b_ih, gates, 4 * HH, BB, 4 * HH, 2 * EE, 0, stream);
        // gates += h @ W_hh^T + b_hh (K = H = 512)
        gemm(h_bf, HH, whh_bf, HH, b_hh, gates, 4 * HH, BB, 4 * HH, HH, 1, stream);

        lstm_cell_kernel<<<(BB * HH) / 256, 256, 0, stream>>>(gates, h, c, h_bf);

        // vocab projection straight into output slice [b, t, :]
        gemm(h_bf, HH, fcw_bf, HH, fc_b, outw + (size_t)t * VV,
             (long long)TT * VV, BB, VV, HH, 0, stream);
    }
}
